// SEIRPhysicsPINN_38276748542127
// MI455X (gfx1250) — matrix operand order repaired and checked by reading, not compiled, not run on hardware
//
#include <hip/hip_runtime.h>
#include <hip/hip_bf16.h>

// ---------------------------------------------------------------------------
// SEIR scan: I_sim[B=128,T=256,N=1024], Pi[N,N].
// Persistent kernel, bf16 WMMA (v_wmma_f32_16x16x32_bf16), f32 accumulate.
// S/E/I state lives in registers for all 255 steps; only I touches memory.
// ---------------------------------------------------------------------------

#define Bdim 128
#define Tdim 256
#define Ndim 1024
#define TN   (Tdim * Ndim)     // 262144, per-b stride in I_sim
#define NBLK 64                // persistent grid size (8 mt x 8 nt-groups)

typedef __attribute__((ext_vector_type(16))) __bf16 v16bf;
typedef __attribute__((ext_vector_type(8)))  __bf16 v8bf;
typedef __attribute__((ext_vector_type(4)))  __bf16 v4bf;
typedef __attribute__((ext_vector_type(8)))  float  v8f;

union AFrag { v16bf v; v8bf h[2]; };

// ---------------------------------------------------------------------------
// Kernel 1: row softmax of pi_logits -> Pi (f32, written to d_out tail)
// ---------------------------------------------------------------------------
__global__ __launch_bounds__(256) void softmax_rows(const float* __restrict__ in,
                                                    float* __restrict__ out) {
    const int row = blockIdx.x;
    const int tid = threadIdx.x;
    __shared__ float red[256];

    const float* r = in + (size_t)row * Ndim;
    float v0 = r[tid], v1 = r[tid + 256], v2 = r[tid + 512], v3 = r[tid + 768];

    float m = fmaxf(fmaxf(v0, v1), fmaxf(v2, v3));
    red[tid] = m;
    __syncthreads();
    for (int s = 128; s > 0; s >>= 1) {
        if (tid < s) red[tid] = fmaxf(red[tid], red[tid + s]);
        __syncthreads();
    }
    m = red[0];
    __syncthreads();

    v0 = __expf(v0 - m); v1 = __expf(v1 - m);
    v2 = __expf(v2 - m); v3 = __expf(v3 - m);
    red[tid] = v0 + v1 + v2 + v3;
    __syncthreads();
    for (int s = 128; s > 0; s >>= 1) {
        if (tid < s) red[tid] += red[tid + s];
        __syncthreads();
    }
    const float inv = 1.0f / red[0];

    float* o = out + (size_t)row * Ndim;
    o[tid] = v0 * inv; o[tid + 256] = v1 * inv;
    o[tid + 512] = v2 * inv; o[tid + 768] = v3 * inv;
}

// ---------------------------------------------------------------------------
// Kernel 2: pack Pi (f32) into bf16 B-fragments.
// Fragment layout (ISA 7.12.2, 16-bit B 32x16, wave32):
//   lane L<16 : column N = L,      K = {0..7,16..23}  (2 per VGPR)
//   lane L>=16: column N = L-16,   K = {8..15,24..31}
// Stored so each lane loads its 16 bf16 values contiguously (32 B).
// Buffer index: ((kt*64 + nt)*32 + lane)*16 + j
// Also zeroes the grid-barrier counter (must happen every call).
// ---------------------------------------------------------------------------
__global__ __launch_bounds__(256) void pack_pi(const float* __restrict__ Pi,
                                               __bf16* __restrict__ packed,
                                               unsigned* __restrict__ ctr) {
    const int idx = blockIdx.x * 256 + threadIdx.x;   // 0 .. N*N-1
    if (idx == 0) *ctr = 0u;
    const int j    = idx & 15;
    const int lane = (idx >> 4) & 31;
    const int tile = idx >> 9;        // kt*64 + nt
    const int nt   = tile & 63;
    const int kt   = tile >> 6;
    const int k = kt * 32 + (lane >> 4) * 8 + (j & 7) + ((j >> 3) << 4);
    const int n = nt * 16 + (lane & 15);
    packed[idx] = (__bf16)Pi[(size_t)k * Ndim + n];
}

// ---------------------------------------------------------------------------
// Grid-wide barrier: monotonic arrive counter in workspace (64 blocks are
// always co-resident). Thread 0 arrives with agent-scope atomic and spins.
// ---------------------------------------------------------------------------
__device__ __forceinline__ void grid_sync(unsigned* ctr, unsigned target) {
    __syncthreads();
    __threadfence();                       // release: make I_t stores visible
    if (threadIdx.x == 0) {
        atomicAdd(ctr, 1u);
        while (__hip_atomic_load(ctr, __ATOMIC_ACQUIRE,
                                 __HIP_MEMORY_SCOPE_AGENT) < target) {
            __builtin_amdgcn_s_sleep(1);
        }
    }
    __syncthreads();
    __threadfence();                       // acquire: don't read stale I_{t}
}

// ---------------------------------------------------------------------------
// Kernel 3: persistent SEIR scan.
// grid = 64 blocks x 256 threads (8 waves). blockIdx: mt = x&7 (b-tile),
// ntg = x>>3; wave w handles n-tile nt = ntg*8 + w. One 16x16 tile per wave.
// Per step: stage I_{t-1}[mt*16..+15, :] into LDS (bf16, padded rows),
// 32 x v_wmma_f32_16x16x32_bf16, register SEIR update, store I_t, grid sync.
// ---------------------------------------------------------------------------
__global__ __launch_bounds__(256) void seir_persistent(
    const float* __restrict__ x_hist,
    const float* __restrict__ beta,
    const float* __restrict__ sigma,
    const float* __restrict__ gamma,
    const __bf16* __restrict__ PiPacked,
    float* __restrict__ Iout,            // d_out: I_sim [B,T,N]
    unsigned* __restrict__ ctr)
{
    __shared__ __bf16 Alds[16][1032];    // padded: 2064 B row stride -> lanes
                                         // 0..15 hit banks 4L..4L+3 (no conflict)
    const int tid  = threadIdx.x;
    const int mt   = blockIdx.x & 7;
    const int ntg  = blockIdx.x >> 3;
    const int wave = tid >> 5;
    const int lane = tid & 31;
    const int nt   = ntg * 8 + wave;
    const int hi   = lane >> 4;          // half-wave select
    const int nl   = lane & 15;
    const int n_g  = nt * 16 + nl;       // this thread's n column

    const float bet = beta[n_g], sig = sigma[n_g], gam = gamma[n_g];

    // Register-resident state for 8 rows: M = hi*8 + r  (C/D VGPR layout)
    float S[8], E[8], I[8];
    int   bg[8];
#pragma unroll
    for (int r = 0; r < 8; ++r) {
        bg[r] = mt * 16 + hi * 8 + r;
        I[r]  = fmaxf(x_hist[(size_t)bg[r] * TN + n_g], 0.0f);   // x_hist[b,0,n]
        S[r]  = 0.99f;
        E[r]  = 0.0f;
        Iout[(size_t)bg[r] * TN + n_g] = I[r];                   // I_sim[b,0,n]
    }

    unsigned bar = 1;
    grid_sync(ctr, NBLK * bar);          // all of I_0 visible device-wide

    for (int t = 1; t < Tdim; ++t) {
        // ---- stage A = I_{t-1} rows [mt*16, +16) x [0,1024) as bf16 in LDS
        const size_t tprev = (size_t)(t - 1) * Ndim;
#pragma unroll 4
        for (int m = 0; m < 16; ++m) {
            const float4 f = *(const float4*)(Iout + (size_t)(mt * 16 + m) * TN
                                                   + tprev + (size_t)tid * 4);
            v4bf p = { (__bf16)f.x, (__bf16)f.y, (__bf16)f.z, (__bf16)f.w };
            *(v4bf*)&Alds[m][tid * 4] = p;
        }
        __syncthreads();

        // ---- GEMM tile: lam[16x16] = I_{t-1}[mt] @ Pi[:, nt]
        v8f acc = {};
        const __bf16* bcol = PiPacked + ((size_t)nt * 32 + lane) * 16;
#pragma unroll 4
        for (int kt = 0; kt < 32; ++kt) {
            // A frag (ISA 16-bit A 16x32): lanes<16: M=lane, K={0..7,16..23};
            // lanes>=16: M=lane-16, K={8..15,24..31}  -> two ds_load_b128
            const int kbase = kt * 32 + hi * 8;
            AFrag a;
            a.h[0] = *(const v8bf*)&Alds[nl][kbase];
            a.h[1] = *(const v8bf*)&Alds[nl][kbase + 16];
            // B frag: one contiguous 32 B load per lane from packed Pi (L2-resident)
            const v16bf bf = *(const v16bf*)(bcol + (size_t)kt * (64 * 512));
            acc = __builtin_amdgcn_wmma_f32_16x16x32_bf16(
                      false, a.v, false, bf, (short)0, acc, false, false);
        }

        // ---- element-wise SEIR update (all state in registers)
#pragma unroll
        for (int r = 0; r < 8; ++r) {
            const float lam = acc[r];
            const float dE  = bet * S[r] * lam - sig * E[r];
            const float dI  = sig * E[r] - gam * I[r];
            E[r] += dE;
            I[r]  = fmaxf(I[r] + dI, 0.0f);
            S[r]  = fmaxf(1.0f - (E[r] + I[r]), 0.01f);
            Iout[(size_t)bg[r] * TN + (size_t)t * Ndim + n_g] = I[r];
        }

        ++bar;
        grid_sync(ctr, NBLK * bar);      // I_t visible before anyone stages it
    }
}

// ---------------------------------------------------------------------------
extern "C" void kernel_launch(void* const* d_in, const int* in_sizes, int n_in,
                              void* d_out, int out_size, void* d_ws, size_t ws_size,
                              hipStream_t stream) {
    const float* x_hist    = (const float*)d_in[0];
    const float* beta      = (const float*)d_in[1];
    const float* sigma     = (const float*)d_in[2];
    const float* gamma     = (const float*)d_in[3];
    const float* pi_logits = (const float*)d_in[4];

    float* out = (float*)d_out;
    float* Pi  = out + (size_t)Bdim * Tdim * Ndim;   // second output, [N,N]

    unsigned* ctr     = (unsigned*)d_ws;
    __bf16*   packed  = (__bf16*)((char*)d_ws + 256); // 2 MB bf16 Pi fragments

    softmax_rows<<<Ndim, 256, 0, stream>>>(pi_logits, Pi);
    pack_pi<<<(Ndim * Ndim) / 256, 256, 0, stream>>>(Pi, packed, ctr);
    seir_persistent<<<NBLK, 256, 0, stream>>>(x_hist, beta, sigma, gamma,
                                              packed, out, ctr);
}
